// LSTM_Classifier_15307263443468
// MI455X (gfx1250) — matrix operand order repaired and checked by reading, not compiled, not run on hardware
//
#include <hip/hip_runtime.h>
#include <hip/hip_bf16.h>

typedef __attribute__((ext_vector_type(16))) _Float16 v16h;
typedef __attribute__((ext_vector_type(8)))  _Float16 v8h;
typedef __attribute__((ext_vector_type(8)))  float    v8f;
typedef __attribute__((ext_vector_type(4)))  int      v4i;

#define B2     8192
#define BATCHN 4096
#define HID    512
#define TSTEPS 20
#define GDIM   2048                 /* 4*HID */
#define KFC    (2 * TSTEPS * HID)   /* 20480 */

// ---------- optional CDNA5 async global->LDS copy (ASYNCcnt-tracked) ----------
#if defined(__has_builtin)
#if __has_builtin(__builtin_amdgcn_global_load_async_to_lds_b128)
#define HAVE_ASYNC_COPY 1
#endif
#endif

#ifdef HAVE_ASYNC_COPY
// Signature (from compiler diagnostic): (v4i as1*, v4i as3*, imm int, imm int)
__device__ __forceinline__ void async_copy16(const _Float16* gsrc, _Float16* ldst) {
  __builtin_amdgcn_global_load_async_to_lds_b128(
      (__attribute__((address_space(1))) v4i*)gsrc,
      (__attribute__((address_space(3))) v4i*)ldst, 0, 0);
}
__device__ __forceinline__ void async_wait() {
  asm volatile("s_wait_asynccnt 0" ::: "memory");
}
#endif

__device__ __forceinline__ float sigmoidf_(float x) {
  x = fminf(fmaxf(x, -30.0f), 30.0f);
  return 1.0f / (1.0f + __expf(-x));
}
__device__ __forceinline__ float tanhf_(float x) {
  x = fminf(fmaxf(x, -15.0f), 15.0f);
  float e = __expf(2.0f * x);
  return (e - 1.0f) / (e + 1.0f);
}

// ---- CDNA5 WMMA fragment loaders (wave32, 16x16x32 f16) ----
// A (16x32, MxK): lanes 0-15: M=lane, K={0..7,16..23}; lanes 16-31: M=lane-16, K={8..15,24..31}
__device__ __forceinline__ v16h load_frag_a(const _Float16* smem, int stride, int mbase, int lane) {
  int m  = mbase + (lane & 15);
  int kb = (lane & 16) ? 8 : 0;
  const _Float16* p = smem + m * stride + kb;
  union { v16h v; v8h h[2]; } u;
  u.h[0] = *(const v8h*)(p);        // K = kb .. kb+7
  u.h[1] = *(const v8h*)(p + 16);   // K = kb+16 .. kb+23
  return u.v;
}
// B (32x16, KxN): lanes 0-15: N=lane, K={0..7,16..23}; lanes 16-31: N=lane-16, K={8..15,24..31}
// Bs stored as [n][k] so each lane reads two runs of 8 contiguous halves.
__device__ __forceinline__ v16h load_frag_b(const _Float16* smem, int stride, int nbase, int lane) {
  int n  = nbase + (lane & 15);
  int kb = (lane & 16) ? 8 : 0;
  const _Float16* p = smem + n * stride + kb;
  union { v16h v; v8h h[2]; } u;
  u.h[0] = *(const v8h*)(p);
  u.h[1] = *(const v8h*)(p + 16);
  return u.v;
}

// ---------------- LSTM step: gates GEMM (WMMA) + fused cell update ----------------
// Block: 64 rows (B2 dim) x 64 hidden cols; 8 waves; wave = 32(M)x16(N) x 4 gates
// (8 WMMA / wave / K-step). Double-buffered LDS, one barrier per K-step.
__global__ __launch_bounds__(256) void lstm_step_kernel(
    const _Float16* __restrict__ hprev, int hstride,
    const _Float16* __restrict__ whh,     // f16 [2048, 512] (gate-major rows, K contiguous)
    const float*    __restrict__ x,       // [B2, T, 2]
    const float*    __restrict__ wih,     // [2048, 2]
    const float*    __restrict__ bias,    // [2048] = b_ih + b_hh
    const float*    __restrict__ cin,     // [B2, 512]
    float*          __restrict__ cout,    // [B2, 512]
    _Float16*       __restrict__ states,  // f16 [B2, T, 512]
    int t)
{
  __shared__ _Float16 As[2][64 * 40];        // 10 KB
  __shared__ _Float16 Bs[2][4][64 * 40];     // 40 KB

  const int tid  = threadIdx.x;
  const int wave = tid >> 5;
  const int lane = tid & 31;
  const int m0 = blockIdx.x * 64;   // row block in B2
  const int n0 = blockIdx.y * 64;   // hidden-col block
  const int mw = (wave & 1) * 32;
  const int nw = (wave >> 1) * 16;

  // per-thread copy mapping: 16-byte chunks (all 16B-aligned: LDS row stride = 80B)
  const int arow = tid >> 2;          // 0..63
  const int akk  = (tid & 3) << 3;    // 0,8,16,24
  const _Float16* aPtr = hprev + (size_t)(m0 + arow) * hstride + akk;
  _Float16* aDst[2] = { &As[0][arow * 40 + akk], &As[1][arow * 40 + akk] };

  const _Float16* bPtr[4];
  _Float16* bDst[2][4];
  #pragma unroll
  for (int it = 0; it < 4; ++it) {
    int c  = tid + (it << 8);         // 0..1023
    int kk = (c & 3) << 3;
    int n  = (c >> 2) & 63;
    int g  = c >> 8;
    bPtr[it]    = whh + (size_t)(g * HID + n0 + n) * HID + kk;
    bDst[0][it] = &Bs[0][g][n * 40 + kk];
    bDst[1][it] = &Bs[1][g][n * 40 + kk];
  }

  v8f acc[4][2] = {};

  auto compute = [&](int cur) {
    v16h a0 = load_frag_a(As[cur], 40, mw, lane);
    v16h a1 = load_frag_a(As[cur], 40, mw + 16, lane);
    #pragma unroll
    for (int g = 0; g < 4; ++g) {
      v16h b = load_frag_b(Bs[cur][g], 40, nw, lane);
      acc[g][0] = __builtin_amdgcn_wmma_f32_16x16x32_f16(
          false, a0, false, b, (short)0, acc[g][0], false, false);
      acc[g][1] = __builtin_amdgcn_wmma_f32_16x16x32_f16(
          false, a1, false, b, (short)0, acc[g][1], false, false);
    }
  };

  const int NK = HID / 32;  // 16
#ifdef HAVE_ASYNC_COPY
  {
    async_copy16(aPtr, aDst[0]);
    #pragma unroll
    for (int it = 0; it < 4; ++it) async_copy16(bPtr[it], bDst[0][it]);
    async_wait();
    __syncthreads();
  }
  for (int k0 = 0; k0 < NK; ++k0) {
    int cur = k0 & 1;
    if (k0 + 1 < NK) {
      int k = (k0 + 1) * 32;
      async_copy16(aPtr + k, aDst[1 - cur]);
      #pragma unroll
      for (int it = 0; it < 4; ++it) async_copy16(bPtr[it] + k, bDst[1 - cur][it]);
    }
    compute(cur);
    async_wait();
    __syncthreads();
  }
#else
  v8h ra;
  v8h rb[4];
  {
    ra = *(const v8h*)(aPtr);
    #pragma unroll
    for (int it = 0; it < 4; ++it) rb[it] = *(const v8h*)(bPtr[it]);
    *(v8h*)aDst[0] = ra;
    #pragma unroll
    for (int it = 0; it < 4; ++it) *(v8h*)bDst[0][it] = rb[it];
    __syncthreads();
  }
  for (int k0 = 0; k0 < NK; ++k0) {
    int cur = k0 & 1;
    if (k0 + 1 < NK) {
      int k = (k0 + 1) * 32;
      ra = *(const v8h*)(aPtr + k);
      #pragma unroll
      for (int it = 0; it < 4; ++it) rb[it] = *(const v8h*)(bPtr[it] + k);
    }
    compute(cur);
    if (k0 + 1 < NK) {
      *(v8h*)aDst[1 - cur] = ra;
      #pragma unroll
      for (int it = 0; it < 4; ++it) *(v8h*)bDst[1 - cur][it] = rb[it];
    }
    __syncthreads();
  }
#endif

  // Fused epilogue: gates -> (c,h). Per-lane column is fixed (N = lane&15).
  const int ncol = n0 + nw + (lane & 15);
  float w0[4], w1[4], bb[4];
  #pragma unroll
  for (int g = 0; g < 4; ++g) {
    int gc = g * HID + ncol;
    w0[g] = wih[gc * 2 + 0];
    w1[g] = wih[gc * 2 + 1];
    bb[g] = bias[gc];
  }
  #pragma unroll
  for (int s = 0; s < 2; ++s) {
    const int mbase = m0 + mw + s * 16 + ((lane & 16) ? 8 : 0);
    #pragma unroll
    for (int e = 0; e < 8; ++e) {
      int r = mbase + e;
      float x0 = x[(size_t)r * (TSTEPS * 2) + t * 2 + 0];
      float x1 = x[(size_t)r * (TSTEPS * 2) + t * 2 + 1];
      float gi = acc[0][s][e] + x0 * w0[0] + x1 * w1[0] + bb[0];
      float gf = acc[1][s][e] + x0 * w0[1] + x1 * w1[1] + bb[1];
      float gg = acc[2][s][e] + x0 * w0[2] + x1 * w1[2] + bb[2];
      float go = acc[3][s][e] + x0 * w0[3] + x1 * w1[3] + bb[3];
      float i_ = sigmoidf_(gi);
      float f_ = sigmoidf_(gf);
      float g_ = tanhf_(gg);
      float o_ = sigmoidf_(go);
      float c_ = f_ * cin[(size_t)r * HID + ncol] + i_ * g_;
      cout[(size_t)r * HID + ncol] = c_;
      float h_ = o_ * tanhf_(c_);
      states[(size_t)r * (TSTEPS * HID) + t * HID + ncol] = (_Float16)h_;
    }
  }
}

// ---------------- fc1: [4096 x 20480] x [20480 x 512] + bias + ReLU ----------------
// Block: 64(M) x 64(N); wave = 32(M)x16(N); double-buffered LDS.
__global__ __launch_bounds__(256) void fc1_kernel(
    const _Float16* __restrict__ states,  // [4096, 20480] f16
    const _Float16* __restrict__ w1,      // [512, 20480] f16
    const float*    __restrict__ b1,      // [512]
    float*          __restrict__ hidden)  // [4096, 512]
{
  __shared__ _Float16 As[2][64 * 40];
  __shared__ _Float16 Bs[2][64 * 40];

  const int tid  = threadIdx.x;
  const int wave = tid >> 5;
  const int lane = tid & 31;
  const int m0 = blockIdx.x * 64;
  const int n0 = blockIdx.y * 64;
  const int mw = (wave & 1) * 32;
  const int nw = (wave >> 1) * 16;

  const int row = tid >> 2;           // 0..63
  const int kk  = (tid & 3) << 3;     // 0,8,16,24
  const _Float16* aPtr = states + (size_t)(m0 + row) * KFC + kk;
  const _Float16* bPtr = w1 + (size_t)(n0 + row) * KFC + kk;
  _Float16* aDst[2] = { &As[0][row * 40 + kk], &As[1][row * 40 + kk] };
  _Float16* bDst[2] = { &Bs[0][row * 40 + kk], &Bs[1][row * 40 + kk] };

  v8f acc[2] = {};

  auto compute = [&](int cur) {
    v16h a0 = load_frag_a(As[cur], 40, mw, lane);
    v16h a1 = load_frag_a(As[cur], 40, mw + 16, lane);
    v16h b  = load_frag_b(Bs[cur], 40, nw, lane);
    acc[0] = __builtin_amdgcn_wmma_f32_16x16x32_f16(
        false, a0, false, b, (short)0, acc[0], false, false);
    acc[1] = __builtin_amdgcn_wmma_f32_16x16x32_f16(
        false, a1, false, b, (short)0, acc[1], false, false);
  };

  const int NK = KFC / 32;  // 640
#ifdef HAVE_ASYNC_COPY
  {
    async_copy16(aPtr, aDst[0]);
    async_copy16(bPtr, bDst[0]);
    async_wait();
    __syncthreads();
  }
  for (int k0 = 0; k0 < NK; ++k0) {
    int cur = k0 & 1;
    if (k0 + 1 < NK) {
      int k = (k0 + 1) * 32;
      async_copy16(aPtr + k, aDst[1 - cur]);
      async_copy16(bPtr + k, bDst[1 - cur]);
    }
    compute(cur);
    async_wait();
    __syncthreads();
  }
#else
  v8h ra, rbv;
  {
    ra  = *(const v8h*)(aPtr);
    rbv = *(const v8h*)(bPtr);
    *(v8h*)aDst[0] = ra;
    *(v8h*)bDst[0] = rbv;
    __syncthreads();
  }
  for (int k0 = 0; k0 < NK; ++k0) {
    int cur = k0 & 1;
    if (k0 + 1 < NK) {
      int k = (k0 + 1) * 32;
      ra  = *(const v8h*)(aPtr + k);
      rbv = *(const v8h*)(bPtr + k);
    }
    compute(cur);
    if (k0 + 1 < NK) {
      *(v8h*)aDst[1 - cur] = ra;
      *(v8h*)bDst[1 - cur] = rbv;
    }
    __syncthreads();
  }
#endif

  const int ncol = n0 + nw + (lane & 15);
  const float bv = b1[ncol];
  #pragma unroll
  for (int s = 0; s < 2; ++s) {
    const int mbase = m0 + mw + s * 16 + ((lane & 16) ? 8 : 0);
    #pragma unroll
    for (int e = 0; e < 8; ++e) {
      float v = acc[s][e] + bv;
      hidden[(size_t)(mbase + e) * HID + ncol] = fmaxf(v, 0.0f);
    }
  }
}

// ---------------- fc2: per-row dot(512) + sigmoid ----------------
__global__ __launch_bounds__(256) void fc2_kernel(
    const float* __restrict__ hidden, const float* __restrict__ w2,
    const float* __restrict__ b2, float* __restrict__ out)
{
  const int wave = threadIdx.x >> 5;
  const int lane = threadIdx.x & 31;
  const int row  = blockIdx.x * 8 + wave;
  const float* hr = hidden + (size_t)row * HID;
  float s = 0.0f;
  #pragma unroll
  for (int i = 0; i < HID / 32; ++i) {
    int j = lane + i * 32;
    s += hr[j] * w2[j];
  }
  #pragma unroll
  for (int off = 16; off > 0; off >>= 1)
    s += __shfl_down(s, off, 32);
  if (lane == 0) out[row] = sigmoidf_(s + b2[0]);
}

// ---------------- prep kernels ----------------
__global__ void f32_to_f16_kernel(const float* __restrict__ src,
                                  _Float16* __restrict__ dst, int n) {
  int i = blockIdx.x * blockDim.x + threadIdx.x;
  if (i < n) dst[i] = (_Float16)src[i];
}
__global__ void bias_combine_kernel(const float* __restrict__ a,
                                    const float* __restrict__ b,
                                    float* __restrict__ o, int n) {
  int i = blockIdx.x * blockDim.x + threadIdx.x;
  if (i < n) o[i] = a[i] + b[i];
}

extern "C" void kernel_launch(void* const* d_in, const int* in_sizes, int n_in,
                              void* d_out, int out_size, void* d_ws, size_t ws_size,
                              hipStream_t stream) {
  (void)in_sizes; (void)n_in; (void)out_size; (void)ws_size;
  const float* x     = (const float*)d_in[0];
  const float* h0    = (const float*)d_in[2];
  const float* c0    = (const float*)d_in[3];
  const float* W_ih  = (const float*)d_in[4];
  const float* W_hh  = (const float*)d_in[5];
  const float* b_ih  = (const float*)d_in[6];
  const float* b_hh  = (const float*)d_in[7];
  const float* fc1_w = (const float*)d_in[8];
  const float* fc1_b = (const float*)d_in[9];
  const float* fc2_w = (const float*)d_in[10];
  const float* fc2_b = (const float*)d_in[11];
  float* out = (float*)d_out;

  char* ws = (char*)d_ws;
  size_t off = 0;
  auto alloc = [&](size_t bytes) -> void* {
    off = (off + 255) & ~(size_t)255;
    void* p = ws + off;
    off += bytes;
    return p;
  };
  _Float16* states = (_Float16*)alloc((size_t)B2 * TSTEPS * HID * 2);  // 168 MB
  _Float16* h0f    = (_Float16*)alloc((size_t)B2 * HID * 2);
  _Float16* whhf   = (_Float16*)alloc((size_t)GDIM * HID * 2);
  _Float16* fc1f   = (_Float16*)alloc((size_t)HID * KFC * 2);
  float*    bias   = (float*)alloc((size_t)GDIM * 4);
  float*    cbuf   = (float*)alloc((size_t)B2 * HID * 4);
  float*    hidden = (float*)alloc((size_t)BATCHN * HID * 4);

  f32_to_f16_kernel<<<(GDIM * HID + 255) / 256, 256, 0, stream>>>(W_hh, whhf, GDIM * HID);
  f32_to_f16_kernel<<<(B2 * HID + 255) / 256, 256, 0, stream>>>(h0, h0f, B2 * HID);
  f32_to_f16_kernel<<<(HID * KFC + 255) / 256, 256, 0, stream>>>(fc1_w, fc1f, HID * KFC);
  bias_combine_kernel<<<(GDIM + 255) / 256, 256, 0, stream>>>(b_ih, b_hh, bias, GDIM);

  for (int t = 0; t < TSTEPS; ++t) {
    const _Float16* hprev = (t == 0) ? h0f : (states + (size_t)(t - 1) * HID);
    int hstride = (t == 0) ? HID : (TSTEPS * HID);
    const float* cin = (t == 0) ? c0 : cbuf;
    dim3 grid(B2 / 64, (GDIM / 4) / 64);  // 128 x 8
    lstm_step_kernel<<<grid, 256, 0, stream>>>(hprev, hstride, whhf, x, W_ih,
                                               bias, cin, cbuf, states, t);
  }

  fc1_kernel<<<dim3(BATCHN / 64, HID / 64), 256, 0, stream>>>(states, fc1f, fc1_b, hidden);
  fc2_kernel<<<BATCHN / 8, 256, 0, stream>>>(hidden, fc2_w, fc2_b, out);
}
